// Linear2D_7361573945573
// MI455X (gfx1250) — matrix operand order repaired and checked by reading, not compiled, not run on hardware
//
#include <hip/hip_runtime.h>

typedef __attribute__((ext_vector_type(16))) _Float16 v16h;
typedef __attribute__((ext_vector_type(8)))  _Float16 v8h;
typedef __attribute__((ext_vector_type(4)))  _Float16 v4h;
typedef __attribute__((ext_vector_type(8)))  float    v8f;
typedef __attribute__((ext_vector_type(4)))  float    v4f;

#define K_DIM   256
#define N_DIM   512
#define M_DIM   (4 * 32 * 32 * 32)   // 131072
#define BLOCK_M 128                  // 8 waves x 16 rows
#define BLOCK_N 32                   // 2 x 16-col WMMA tiles per wave
#define KP      264                  // padded LDS K-stride in halves: 528B rows ->
                                     // 4-bank rotation per row, 16B aligned for b128

// D = x * W^T + b with fp32 I/O via split-precision f16 WMMA:
//   A*B ~= Ah*Bh + Ah*Bl + Al*Bh   (f32 accumulation, ~fp32 accuracy)
//
// Grid: blockIdx.x = N-block (FAST dim, 16 blocks) so consecutive blocks share
// the same 128 x-rows -> x is read from HBM once and hit in L2 thereafter.
__global__ __launch_bounds__(256) void linear2d_wmma_kernel(
    const float* __restrict__ x, const float* __restrict__ W,
    const float* __restrict__ bias, float* __restrict__ out)
{
    __shared__ _Float16 ldsH[BLOCK_N * KP];
    __shared__ _Float16 ldsL[BLOCK_N * KP];

    const int t  = threadIdx.x;
    const int n0 = blockIdx.x * BLOCK_N;   // N is the fast grid dimension

    // ---- cooperative W tile load: fp32 -> (hi, lo) f16 split into LDS ----
    // 32 rows x 256 cols = 2048 float4 units, 8 per thread, coalesced.
    #pragma unroll
    for (int i = 0; i < (BLOCK_N * K_DIM / 4) / 256; ++i) {
        const int u  = i * 256 + t;
        const int n  = u >> 6;            // 64 float4 per K-row
        const int kq = (u & 63) << 2;     // k offset in floats
        v4f w = *(const v4f*)(W + (n0 + n) * K_DIM + kq);
        v4h h, l;
        #pragma unroll
        for (int j = 0; j < 4; ++j) {
            _Float16 hh = (_Float16)w[j];
            h[j] = hh;
            l[j] = (_Float16)(w[j] - (float)hh);
        }
        *(v4h*)(&ldsH[n * KP + kq]) = h;
        *(v4h*)(&ldsL[n * KP + kq]) = l;
    }
    __syncthreads();

    const int wave = t >> 5;
    const int lane = t & 31;
    const int g    = lane >> 4;   // K half-group
    const int r    = lane & 15;   // A row within tile / B column within tile
    const int m0   = blockIdx.y * BLOCK_M + wave * 16;

    const float* arow = x + (m0 + r) * K_DIM;

    v8f acc0 = {};
    v8f acc1 = {};

    #pragma unroll
    for (int kk = 0; kk < K_DIM; kk += 32) {
        // A tile 16x32 (ISA A-layout): lane holds row r;
        // elems 0..7 -> K=kb..kb+7, elems 8..15 -> K=kb+16..kb+23, kb=kk+8g.
        const int kb = kk + g * 8;
        v4f a0 = *(const v4f*)(arow + kb);
        v4f a1 = *(const v4f*)(arow + kb + 4);
        v4f a2 = *(const v4f*)(arow + kb + 16);
        v4f a3 = *(const v4f*)(arow + kb + 20);

        v16h ah, al;
        #pragma unroll
        for (int j = 0; j < 4; ++j) {
            _Float16 h;
            h = (_Float16)a0[j]; ah[j]      = h; al[j]      = (_Float16)(a0[j] - (float)h);
            h = (_Float16)a1[j]; ah[4 + j]  = h; al[4 + j]  = (_Float16)(a1[j] - (float)h);
            h = (_Float16)a2[j]; ah[8 + j]  = h; al[8 + j]  = (_Float16)(a2[j] - (float)h);
            h = (_Float16)a3[j]; ah[12 + j] = h; al[12 + j] = (_Float16)(a3[j] - (float)h);
        }

        // B tile 32x16 (matches A): lane holds col r;
        // elems 0..7 -> K=kbB..kbB+7, elems 8..15 -> K=kbB+16..kbB+23, kbB=kk+8g.
        const int kbB = kk + g * 8;
        #pragma unroll
        for (int tt = 0; tt < 2; ++tt) {
            const int brow = (tt * 16 + r) * KP + kbB;
            v8h bh0 = *(const v8h*)(&ldsH[brow]);
            v8h bh1 = *(const v8h*)(&ldsH[brow + 16]);
            v8h bl0 = *(const v8h*)(&ldsL[brow]);
            v8h bl1 = *(const v8h*)(&ldsL[brow + 16]);
            v16h bh = __builtin_shufflevector(bh0, bh1,
                       0,1,2,3,4,5,6,7,8,9,10,11,12,13,14,15);
            v16h bl = __builtin_shufflevector(bl0, bl1,
                       0,1,2,3,4,5,6,7,8,9,10,11,12,13,14,15);

            v8f& acc = tt ? acc1 : acc0;
            // correction terms first, leading term last
            acc = __builtin_amdgcn_wmma_f32_16x16x32_f16(
                      false, al, false, bh, (short)0, acc, false, false);
            acc = __builtin_amdgcn_wmma_f32_16x16x32_f16(
                      false, ah, false, bl, (short)0, acc, false, false);
            acc = __builtin_amdgcn_wmma_f32_16x16x32_f16(
                      false, ah, false, bh, (short)0, acc, false, false);
        }
    }

    // ---- epilogue: bias add + non-temporal store per C/D VGPR layout ----
    // VGPR i: lanes 0-15 -> M=m0+i, lanes 16-31 -> M=m0+8+i; N = r.
    // out is write-once: NT stores keep x/W resident in L2.
    #pragma unroll
    for (int tt = 0; tt < 2; ++tt) {
        const int   col = n0 + tt * 16 + r;
        const float bv  = bias[col];
        const v8f&  acc = tt ? acc1 : acc0;
        #pragma unroll
        for (int i = 0; i < 8; ++i) {
            const int row = m0 + g * 8 + i;
            __builtin_nontemporal_store(acc[i] + bv, &out[row * N_DIM + col]);
        }
    }
}

extern "C" void kernel_launch(void* const* d_in, const int* in_sizes, int n_in,
                              void* d_out, int out_size, void* d_ws, size_t ws_size,
                              hipStream_t stream) {
    (void)in_sizes; (void)n_in; (void)d_ws; (void)ws_size; (void)out_size;
    const float* x    = (const float*)d_in[0];
    const float* W    = (const float*)d_in[1];
    const float* bias = (const float*)d_in[2];
    float*       out  = (float*)d_out;

    // N-blocks in x (fast) so blocks sharing x-rows run back-to-back -> L2 reuse.
    dim3 grid(N_DIM / BLOCK_N, M_DIM / BLOCK_M);   // (16, 1024)
    linear2d_wmma_kernel<<<grid, 256, 0, stream>>>(x, W, bias, out);
}
